// LSTMblock_31044023615611
// MI455X (gfx1250) — matrix operand order repaired and checked by reading, not compiled, not run on hardware
//
#include <hip/hip_runtime.h>
#include <hip/hip_bf16.h>

typedef _Float16 half4v __attribute__((ext_vector_type(4)));
typedef _Float16 half8v __attribute__((ext_vector_type(8)));
typedef _Float16 v16h   __attribute__((ext_vector_type(16)));
typedef float    v8f    __attribute__((ext_vector_type(8)));

#define N_HEADS 8
#define NLAYERS 2
#define HDIM    64
#define FDIM    64
#define TLEN    12
#define NNODES  16384
#define GDIM    256   /* 4*H gate rows */
#define KDIM    128   /* F + H concat K */
#define WAVES   4
#define BLOCK   (WAVES * 32)

__device__ __forceinline__ float sigmoidf_(float x) {
    return 1.0f / (1.0f + __expf(-x));
}

// One wave owns 16 nodes for one head; runs the full T-step scan with the
// concatenated weight matrix [W_ih | W_hh] resident in LDS as f16.
__global__ __launch_bounds__(BLOCK) void lstm_layer_kernel(
    const float* __restrict__ x,       // (T,N,F) f32; only used when layer==0
    _Float16* __restrict__ hs,         // (T,HEADS,N,H) f16; layer1 reads+writes in place
    const float* __restrict__ Wih,     // (L,HEADS,4H,H)
    const float* __restrict__ Whh,     // (L,HEADS,4H,H)
    const float* __restrict__ bih,     // (L,HEADS,4H)
    const float* __restrict__ bhh,     // (L,HEADS,4H)
    const float* __restrict__ h0,      // (HEADS,L,N,H)
    const float* __restrict__ c0,      // (HEADS,L,N,H)
    int layer)
{
    __shared__ __align__(32) _Float16 Wsh[GDIM * KDIM];       // 64 KB: [row][k], k<64 = W_ih, k>=64 = W_hh
    __shared__ __align__(16) float    Bsh[GDIM];              // combined bias
    __shared__ __align__(32) _Float16 Ash[WAVES][16 * KDIM];  // per-wave A staging [m][k]

    const int tid  = threadIdx.x;
    const int wv   = tid >> 5;
    const int lane = tid & 31;
    const int hf   = lane >> 4;    // half of wave (0/1)
    const int ln   = lane & 15;
    const int head = blockIdx.y;
    const int nodeBase = blockIdx.x * (16 * WAVES) + wv * 16;

    // ---- cooperative weight + bias load (f32 -> f16 LDS) ----
    const size_t wbase = ((size_t)layer * N_HEADS + head) * (size_t)(GDIM * HDIM);
    for (int idx = tid; idx < GDIM * HDIM; idx += BLOCK) {
        int r = idx >> 6, c = idx & 63;
        Wsh[r * KDIM + c]      = (_Float16)Wih[wbase + idx];
        Wsh[r * KDIM + 64 + c] = (_Float16)Whh[wbase + idx];
    }
    const size_t bbase = ((size_t)layer * N_HEADS + head) * GDIM;
    for (int idx = tid; idx < GDIM; idx += BLOCK)
        Bsh[idx] = bih[bbase + idx] + bhh[bbase + idx];
    __syncthreads();

    // ---- initial h/c state in C-matrix layout: lane: n=ln, reg r: m=r+8*hf ----
    float hst[4][8], cst[4][8];
    {
        const size_t sb = ((size_t)head * NLAYERS + layer) * (size_t)NNODES;
#pragma unroll
        for (int j = 0; j < 4; ++j)
#pragma unroll
            for (int r = 0; r < 8; ++r) {
                size_t node = (size_t)nodeBase + r + 8 * hf;
                size_t off  = (sb + node) * HDIM + 16 * j + ln;
                hst[j][r] = h0[off];
                cst[j][r] = c0[off];
            }
    }

    for (int t = 0; t < TLEN; ++t) {
        // ---- stage this step's input into A cols [0,64) ----
        {
            const int m  = lane >> 1;
            const int cb = (lane & 1) * 32;
            if (layer == 0) {
                const float4* xr = (const float4*)(x + ((size_t)t * NNODES + nodeBase + m) * FDIM + cb);
#pragma unroll
                for (int i = 0; i < 8; ++i) {
                    float4 v = xr[i];
                    half4v hv = {(_Float16)v.x, (_Float16)v.y, (_Float16)v.z, (_Float16)v.w};
                    *(half4v*)&Ash[wv][m * KDIM + cb + 4 * i] = hv;
                }
            } else {
                const half8v* hr = (const half8v*)(hs +
                    (((size_t)t * N_HEADS + head) * NNODES + nodeBase + m) * HDIM + cb);
#pragma unroll
                for (int i = 0; i < 4; ++i)
                    *(half8v*)&Ash[wv][m * KDIM + cb + 8 * i] = hr[i];
            }
        }
        // ---- stage recurrent h into A cols [64,128) (C-layout -> row-major) ----
#pragma unroll
        for (int j = 0; j < 4; ++j)
#pragma unroll
            for (int r = 0; r < 8; ++r)
                Ash[wv][(r + 8 * hf) * KDIM + 64 + 16 * j + ln] = (_Float16)hst[j][r];

        // ---- load the 4 A fragments (K chunks of 32), reused for all 16 tiles ----
        // A 16x32 f16 layout: lane(ln,hf): VGPR0-3 = K[kc*32+hf*8 .. +7], VGPR4-7 = +16
        v16h afr[4];
        const _Float16* arow = &Ash[wv][(size_t)ln * KDIM];
#pragma unroll
        for (int kc = 0; kc < 4; ++kc) {
            const _Float16* p = arow + kc * 32 + hf * 8;
            half8v lo = *(const half8v*)p;
            half8v hi = *(const half8v*)(p + 16);
            afr[kc] = __builtin_shufflevector(lo, hi, 0,1,2,3,4,5,6,7,8,9,10,11,12,13,14,15);
        }

        // ---- gates = [x|h] @ Wcat^T, processed per 16-col group j with i/f/g/o together ----
#pragma unroll
        for (int j = 0; j < 4; ++j) {
            v8f acc[4] = {};  // i, f, g, o accumulators (16x16 f32 tiles)
#pragma unroll
            for (int gg = 0; gg < 4; ++gg) {
                const int row = gg * 64 + j * 16 + ln;  // output gate column owned by this lane
#pragma unroll
                for (int kc = 0; kc < 4; ++kc) {
                    // B 32x16 f16 layout: lane(ln,hf): K[kc*32+hf*8 .. +7], then +16
                    const _Float16* bp = &Wsh[row * KDIM + kc * 32 + hf * 8];
                    const v16h b = __builtin_shufflevector(*(const half8v*)bp, *(const half8v*)(bp + 16), 0,1,2,3,4,5,6,7,8,9,10,11,12,13,14,15);
                    acc[gg] = __builtin_amdgcn_wmma_f32_16x16x32_f16(
                        false, afr[kc], false, b, (short)0, acc[gg], false, false);
                }
            }
            const float bi = Bsh[  0 + 16 * j + ln];
            const float bf = Bsh[ 64 + 16 * j + ln];
            const float bg = Bsh[128 + 16 * j + ln];
            const float bo = Bsh[192 + 16 * j + ln];
#pragma unroll
            for (int r = 0; r < 8; ++r) {
                float iv = sigmoidf_(acc[0][r] + bi);
                float fv = sigmoidf_(acc[1][r] + bf);
                float gv = tanhf(acc[2][r] + bg);
                float ov = sigmoidf_(acc[3][r] + bo);
                float cc = fv * cst[j][r] + iv * gv;
                cst[j][r] = cc;
                float hv = ov * tanhf(cc);
                hst[j][r] = hv;
                size_t node = (size_t)nodeBase + r + 8 * hf;
                hs[(((size_t)t * N_HEADS + head) * NNODES + node) * HDIM + 16 * j + ln] =
                    (_Float16)hv;
            }
        }
    }
}

// out[o,n,h] = conv_b[o] + sum_{head,t} conv_w[o, head*T+t] * hs[t,head,n,h]
__global__ __launch_bounds__(256) void conv_mix_kernel(
    const _Float16* __restrict__ hs, const float* __restrict__ cw,
    const float* __restrict__ cb, float* __restrict__ out)
{
    __shared__ float Wc[TLEN * N_HEADS * TLEN];  // 12 x 96
    for (int i = threadIdx.x; i < TLEN * N_HEADS * TLEN; i += 256) Wc[i] = cw[i];
    __syncthreads();

    size_t gid = (size_t)blockIdx.x * 256 + threadIdx.x;  // over N*H
    int node = (int)(gid >> 6);
    int col  = (int)(gid & 63);

    float acc[TLEN];
#pragma unroll
    for (int o = 0; o < TLEN; ++o) acc[o] = cb[o];

    for (int hd = 0; hd < N_HEADS; ++hd)
#pragma unroll
        for (int t = 0; t < TLEN; ++t) {
            float v = (float)hs[(((size_t)t * N_HEADS + hd) * NNODES + node) * HDIM + col];
            int c = hd * TLEN + t;
#pragma unroll
            for (int o = 0; o < TLEN; ++o)
                acc[o] += Wc[o * (N_HEADS * TLEN) + c] * v;
        }
#pragma unroll
    for (int o = 0; o < TLEN; ++o)
        out[((size_t)o * NNODES + node) * HDIM + col] = acc[o];
}

extern "C" void kernel_launch(void* const* d_in, const int* in_sizes, int n_in,
                              void* d_out, int out_size, void* d_ws, size_t ws_size,
                              hipStream_t stream) {
    (void)in_sizes; (void)n_in; (void)out_size; (void)ws_size;
    const float* x   = (const float*)d_in[0];
    const float* Wih = (const float*)d_in[1];
    const float* Whh = (const float*)d_in[2];
    const float* bih = (const float*)d_in[3];
    const float* bhh = (const float*)d_in[4];
    const float* h0  = (const float*)d_in[5];
    const float* c0  = (const float*)d_in[6];
    const float* cw  = (const float*)d_in[7];
    const float* cb  = (const float*)d_in[8];
    float* out = (float*)d_out;
    _Float16* hs = (_Float16*)d_ws;  // (T,HEADS,N,H) f16 = 192 MB scratch

    dim3 grid(NNODES / (16 * WAVES), N_HEADS);
    lstm_layer_kernel<<<grid, BLOCK, 0, stream>>>(x, hs, Wih, Whh, bih, bhh, h0, c0, 0);
    lstm_layer_kernel<<<grid, BLOCK, 0, stream>>>(nullptr, hs, Wih, Whh, bih, bhh, h0, c0, 1);
    conv_mix_kernel<<<(NNODES * HDIM) / 256, 256, 0, stream>>>(hs, cw, cb, out);
}
